// FactorGraphLayer_40235253629273
// MI455X (gfx1250) — compile-verified, matrix operand order checked by reading, not run on hardware
//
#include <hip/hip_runtime.h>

// ---------------------------------------------------------------------------
// FactorGraph message-passing layer for MI455X (gfx1250, wave32, WMMA).
// D = 64. Each wave processes a 16-row tile: A = 16x128 (concat features),
// B = 128x64 (weights, pre-swizzled to bf16 fragments), C = 16x64 f32 accum
// via v_wmma_f32_16x16x32_bf16. Memory-bound workload: gathers/scatters are
// the roofline term, so hot-path address math is 32-bit, tail guards are
// scalarized (readfirstlane) wave-uniform branches.
// ---------------------------------------------------------------------------

typedef __attribute__((ext_vector_type(16))) __bf16 v16bf;
typedef __attribute__((ext_vector_type(8)))  float  v8f;

union BF16Frag {
    v16bf        v;
    unsigned int d[8];   // dword j = elements {2j (lo), 2j+1 (hi)}
    uint4        q[2];
};

// K-local offset inside a 16x32 bf16 fragment for dword j, lane-half `hi`
// (ISA 7.12.2, 16-bit A layout; B uses the mirrored layout = column-major W).
__device__ __forceinline__ int klocal(int j, int hi) {
    return ((j & 3) << 1) + ((j >> 2) << 4) + (hi << 3);
}

// Native f32 -> bf16 (RNE); lowers to v_cvt_pk_bf16_f32 on gfx1250.
__device__ __forceinline__ void put_pair(BF16Frag& F, int j, float lo, float hi) {
    F.v[2 * j]     = (__bf16)lo;
    F.v[2 * j + 1] = (__bf16)hi;
}

// ---------------------------------------------------------------------------
// Zero-fill (aggregation buffers must be zeroed deterministically each call).
// ---------------------------------------------------------------------------
__global__ void zero_kernel(float* __restrict__ p, long long n) {
    long long i = (long long)blockIdx.x * blockDim.x + threadIdx.x;
    long long stride = (long long)gridDim.x * blockDim.x;
    for (; i < n; i += stride) p[i] = 0.0f;
}

// ---------------------------------------------------------------------------
// Pre-swizzle one [128][64] fp32 weight matrix into bf16 B-fragments.
// Output layout: pre[((tile)*32 + lane)*8 + j], tile = nc*4 + kc.
// Each lane's 8 dwords are the exact v16bf register image for that tile.
// ---------------------------------------------------------------------------
__global__ void preswizzle_kernel(const float* __restrict__ W,
                                  unsigned int* __restrict__ pre) {
    const int t = blockIdx.x * blockDim.x + threadIdx.x;  // (tile, lane)
    if (t >= 16 * 32) return;
    const int tile = t >> 5;
    const int lane = t & 31;
    const int nc = tile >> 2;
    const int kc = tile & 3;
    const int r  = lane & 15;
    const int hi = lane >> 4;
    const int n  = nc * 16 + r;

    BF16Frag F;
#pragma unroll
    for (int j = 0; j < 8; ++j) {
        const int kg = kc * 32 + klocal(j, hi);
        put_pair(F, j, W[kg * 64 + n], W[(kg + 1) * 64 + n]);
    }
    uint4* out = (uint4*)(pre + (unsigned)t * 8u);
    out[0] = F.q[0];
    out[1] = F.q[1];
}

// ---------------------------------------------------------------------------
// Edge message MLP + fused scatter-add:
//   y = relu(concat(nodeA[idxA[e]], nodeB[idxB[e]]) @ W + b)
//   aggr[idxOut[e]] += y        (atomic f32)
// One wave per 16 edges. Wpre = pre-swizzled bf16 fragments (4096 dwords).
// ---------------------------------------------------------------------------
__global__ __launch_bounds__(256) void edge_msg_kernel(
    const float* __restrict__ nodeA, const int* __restrict__ idxA,
    const float* __restrict__ nodeB, const int* __restrict__ idxB,
    const unsigned int* __restrict__ Wpre,
    const float* __restrict__ bias,   // [64]
    float* __restrict__ aggr,         // [*][64]
    const int* __restrict__ idxOut,
    int ntiles, int nrows)
{
    // Scalarize the wave id so all uniform control flow uses SALU branches.
    const int wave = __builtin_amdgcn_readfirstlane(
        (int)((blockIdx.x * blockDim.x + threadIdx.x) >> 5));
    if (wave >= ntiles) return;                      // s_cmp + s_cbranch
    const int lane = threadIdx.x & 31;
    const int r    = lane & 15;
    const int hi   = lane >> 4;
    const int base = wave * 16;                      // SGPR
    const bool full = (base + 16) <= nrows;          // SCC-computable

    int myrow = base + r;
    if (!full && myrow >= nrows) myrow = nrows - 1;  // clamp gathers only
    const int ia = idxA[myrow];
    const int ib = idxB[myrow];
    const float* pa = nodeA + (unsigned)ia * 64u;
    const float* pb = nodeB + (unsigned)ib * 64u;

    // ---- A fragments: 4 x (16x32 bf16) covering K = 0..127
    BF16Frag A[4];
#pragma unroll
    for (int kc = 0; kc < 4; ++kc) {
        const float* p = (kc < 2) ? (pa + kc * 32) : (pb + (kc - 2) * 32);
#pragma unroll
        for (int j = 0; j < 8; ++j) {
            const float2 f = *(const float2*)(p + klocal(j, hi));
            put_pair(A[kc], j, f.x, f.y);
        }
    }

    // ---- scatter row indices for the 8 output rows this lane-half owns
    int orow[8];
    if (full) {
#pragma unroll
        for (int v = 0; v < 8; ++v) orow[v] = idxOut[base + v + hi * 8];
    } else {
#pragma unroll
        for (int v = 0; v < 8; ++v) {
            const int rr = base + v + hi * 8;
            orow[v] = (rr < nrows) ? idxOut[rr] : -1;
        }
    }

    // ---- 4 output column tiles (N = 0..63)
#pragma unroll
    for (int nc = 0; nc < 4; ++nc) {
        const int n = nc * 16 + r;
        v8f acc = {};
#pragma unroll
        for (int kc = 0; kc < 4; ++kc) {
            const int tile = nc * 4 + kc;
            const uint4* bp = (const uint4*)(Wpre + (unsigned)((tile * 32 + lane) * 8));
            BF16Frag B;
            B.q[0] = bp[0];
            B.q[1] = bp[1];
            acc = __builtin_amdgcn_wmma_f32_16x16x32_bf16(
                      false, A[kc].v, false, B.v, (short)0, acc, false, false);
        }
        const float bn = bias[n];
        if (full) {                                   // uniform fast path
#pragma unroll
            for (int v = 0; v < 8; ++v) {
                float y = acc[v] + bn;
                y = y > 0.0f ? y : 0.0f;
                atomicAdd(aggr + ((unsigned)orow[v] * 64u + (unsigned)n), y);
            }
        } else {
#pragma unroll
            for (int v = 0; v < 8; ++v) {
                if (orow[v] >= 0) {
                    float y = acc[v] + bn;
                    y = y > 0.0f ? y : 0.0f;
                    atomicAdd(aggr + ((unsigned)orow[v] * 64u + (unsigned)n), y);
                }
            }
        }
    }
}

// ---------------------------------------------------------------------------
// Node combine MLP:
//   out[i] = (RESIDUAL ? nodeX[i] : 0) + relu(concat(nodeX[i], aggr[i]) @ W + b)
// One wave per 16 nodes (rows are contiguous, no index gather).
// ---------------------------------------------------------------------------
template <bool RESIDUAL>
__global__ __launch_bounds__(256) void combine_kernel(
    const float* __restrict__ nodeX,  // [N][64]
    const float* __restrict__ aggr,   // [N][64]
    const unsigned int* __restrict__ Wpre,
    const float* __restrict__ bias,   // [64]
    float* __restrict__ out,          // [N][64]
    int ntiles, int nrows)
{
    const int wave = __builtin_amdgcn_readfirstlane(
        (int)((blockIdx.x * blockDim.x + threadIdx.x) >> 5));
    if (wave >= ntiles) return;
    const int lane = threadIdx.x & 31;
    const int r    = lane & 15;
    const int hi   = lane >> 4;
    const int base = wave * 16;                      // SGPR
    const bool full = (base + 16) <= nrows;

    int myrow = base + r;
    if (!full && myrow >= nrows) myrow = nrows - 1;
    const float* pa = nodeX + (unsigned)myrow * 64u;
    const float* pb = aggr  + (unsigned)myrow * 64u;

    BF16Frag A[4];
#pragma unroll
    for (int kc = 0; kc < 4; ++kc) {
        const float* p = (kc < 2) ? (pa + kc * 32) : (pb + (kc - 2) * 32);
#pragma unroll
        for (int j = 0; j < 8; ++j) {
            const float2 f = *(const float2*)(p + klocal(j, hi));
            put_pair(A[kc], j, f.x, f.y);
        }
    }

#pragma unroll
    for (int nc = 0; nc < 4; ++nc) {
        const int n = nc * 16 + r;
        v8f acc = {};
#pragma unroll
        for (int kc = 0; kc < 4; ++kc) {
            const int tile = nc * 4 + kc;
            const uint4* bp = (const uint4*)(Wpre + (unsigned)((tile * 32 + lane) * 8));
            BF16Frag B;
            B.q[0] = bp[0];
            B.q[1] = bp[1];
            acc = __builtin_amdgcn_wmma_f32_16x16x32_bf16(
                      false, A[kc].v, false, B.v, (short)0, acc, false, false);
        }
        const float bn = bias[n];
        if (full) {                                   // uniform fast path
#pragma unroll
            for (int v = 0; v < 8; ++v) {
                const unsigned off = (unsigned)(base + v + hi * 8) * 64u + (unsigned)n;
                float y = acc[v] + bn;
                y = y > 0.0f ? y : 0.0f;
                if (RESIDUAL) y += nodeX[off];
                out[off] = y;
            }
        } else {
#pragma unroll
            for (int v = 0; v < 8; ++v) {
                const int rr = base + v + hi * 8;
                if (rr < nrows) {
                    const unsigned off = (unsigned)rr * 64u + (unsigned)n;
                    float y = acc[v] + bn;
                    y = y > 0.0f ? y : 0.0f;
                    if (RESIDUAL) y += nodeX[off];
                    out[off] = y;
                }
            }
        }
    }
}

// ---------------------------------------------------------------------------
// Launch: zero aggr, pre-swizzle weights, then v2f-msg, f-combine,
// f2v-msg, v-combine.
// ---------------------------------------------------------------------------
extern "C" void kernel_launch(void* const* d_in, const int* in_sizes, int n_in,
                              void* d_out, int out_size, void* d_ws, size_t ws_size,
                              hipStream_t stream) {
    const float* variables = (const float*)d_in[0];
    const float* factors   = (const float*)d_in[1];
    const int*   edge_idx  = (const int*)d_in[2];
    // d_in[3] edge_attr, d_in[4] batch_idx: unused by the reference math
    const float* v2f_msg_W = (const float*)d_in[5];
    const float* v2f_msg_b = (const float*)d_in[6];
    const float* v2f_cmb_W = (const float*)d_in[7];
    const float* v2f_cmb_b = (const float*)d_in[8];
    const float* f2v_msg_W = (const float*)d_in[9];
    const float* f2v_msg_b = (const float*)d_in[10];
    const float* f2v_cmb_W = (const float*)d_in[11];
    const float* f2v_cmb_b = (const float*)d_in[12];

    const int NVAR  = in_sizes[0] / 64;
    const int NFAC  = in_sizes[1] / 64;
    const int NEDGE = in_sizes[2] / 2;
    const int* src = edge_idx;          // edge_index[0]
    const int* dst = edge_idx + NEDGE;  // edge_index[1]

    float* out_v = (float*)d_out;                       // new_v: NVAR x 64
    float* out_f = out_v + (long long)NVAR * 64;        // new_f: NFAC x 64

    // workspace layout: aggr_f | aggr_v | 4 pre-swizzled weight blocks
    float* aggr_f = (float*)d_ws;                       // NFAC x 64
    float* aggr_v = aggr_f + (long long)NFAC * 64;      // NVAR x 64
    unsigned int* wpre = (unsigned int*)(aggr_v + (long long)NVAR * 64);
    const int WPRE_DW = 16 * 32 * 8;                    // 4096 dwords / matrix
    unsigned int* pre_v2f_msg = wpre + 0 * WPRE_DW;
    unsigned int* pre_v2f_cmb = wpre + 1 * WPRE_DW;
    unsigned int* pre_f2v_msg = wpre + 2 * WPRE_DW;
    unsigned int* pre_f2v_cmb = wpre + 3 * WPRE_DW;

    const long long nzero = (long long)(NFAC + NVAR) * 64;
    zero_kernel<<<2048, 256, 0, stream>>>(aggr_f, nzero);

    preswizzle_kernel<<<2, 256, 0, stream>>>(v2f_msg_W, pre_v2f_msg);
    preswizzle_kernel<<<2, 256, 0, stream>>>(v2f_cmb_W, pre_v2f_cmb);
    preswizzle_kernel<<<2, 256, 0, stream>>>(f2v_msg_W, pre_f2v_msg);
    preswizzle_kernel<<<2, 256, 0, stream>>>(f2v_cmb_W, pre_f2v_cmb);

    const int tilesE = (NEDGE + 15) / 16;
    const int tilesF = (NFAC  + 15) / 16;
    const int tilesV = (NVAR  + 15) / 16;
    const int WPB = 256 / 32;  // waves per block

    // Phase 1: variable -> factor messages, scatter-add into aggr_f
    edge_msg_kernel<<<(tilesE + WPB - 1) / WPB, 256, 0, stream>>>(
        factors, dst, variables, src, pre_v2f_msg, v2f_msg_b,
        aggr_f, dst, tilesE, NEDGE);

    // Phase 2: new_f = relu([factors | aggr_f] @ W + b)
    combine_kernel<false><<<(tilesF + WPB - 1) / WPB, 256, 0, stream>>>(
        factors, aggr_f, pre_v2f_cmb, v2f_cmb_b, out_f, tilesF, NFAC);

    // Phase 3: factor -> variable messages (uses new_f), scatter into aggr_v
    edge_msg_kernel<<<(tilesE + WPB - 1) / WPB, 256, 0, stream>>>(
        variables, src, out_f, dst, pre_f2v_msg, f2v_msg_b,
        aggr_v, src, tilesE, NEDGE);

    // Phase 4: new_v = variables + relu([variables | aggr_v] @ W + b)
    combine_kernel<true><<<(tilesV + WPB - 1) / WPB, 256, 0, stream>>>(
        variables, aggr_v, pre_f2v_cmb, f2v_cmb_b, out_v, tilesV, NVAR);
}
